// ScaledDotProductAttention_23476291240049
// MI455X (gfx1250) — matrix operand order repaired and checked by reading, not compiled, not run on hardware
//
#include <hip/hip_runtime.h>

typedef __attribute__((ext_vector_type(16))) _Float16 v16h;
typedef __attribute__((ext_vector_type(8)))  float    v8f;
typedef __attribute__((ext_vector_type(4)))  float    v4f;

static constexpr int S_LEN  = 1024;
static constexpr int D_DIM  = 64;
static constexpr int ROW_PAD = 1028;          // floats per score row in LDS (bank-conflict padding)
static constexpr float INV_TEMP = 1.0f / 8.0f;
static constexpr float NEG_BIG  = -1e30f;

__global__ __launch_bounds__(32)
void sdpa_wmma_kernel(const float* __restrict__ q,
                      const float* __restrict__ k,
                      const float* __restrict__ v,
                      const float* __restrict__ factor,
                      const int*   __restrict__ mask,
                      float* __restrict__ out,
                      float* __restrict__ attn)
{
    extern __shared__ float smem[];
    float* sP   = smem;                        // 16 * ROW_PAD  scores -> exp values
    float* sV   = smem + 16 * ROW_PAD;         // 32 * 64       staged V tile
    float* sRed = sV + 32 * 64;                // 32            cross-lane partials
    float* sInv = sRed + 32;                   // 16            1/sum per row

    const int lane = threadIdx.x;              // 0..31 (wave32)
    const int bh   = blockIdx.y;               // 0..63 (B*H)
    const int b    = bh >> 4;                  // H = 16
    const int row0 = blockIdx.x * 16;

    const int nIdx = lane & 15;
    const int hi   = lane >> 4;

    const float* qh = q + (size_t)bh * S_LEN * D_DIM;
    const float* kh = k + (size_t)bh * S_LEN * D_DIM;
    const float* vh = v + (size_t)bh * S_LEN * D_DIM;
    const int*   mh = mask + (size_t)b * S_LEN * S_LEN;
    float* outh  = out  + (size_t)bh * S_LEN * D_DIM;
    float* attnh = attn + (size_t)bh * S_LEN * S_LEN;

    // ---- Q A-fragments, pre-scaled by factor/temperature, f32 -> f16 ----
    // A operand, contiguous k order used here (B matches): lane m = lane&15; elem j -> K = 16*hi + j
    v16h aQ0, aQ1;
    {
        const int   s     = row0 + nIdx;
        const float scale = factor[(size_t)bh * S_LEN + s] * INV_TEMP;
        const float* qrow = qh + (size_t)s * D_DIM;
        const int d0 = 16 * hi;
        #pragma unroll
        for (int j = 0; j < 8; ++j) {
            aQ0[j]     = (_Float16)(qrow[d0 + j]      * scale);
            aQ0[8 + j] = (_Float16)(qrow[d0 + 8 + j] * scale);
            aQ1[j]     = (_Float16)(qrow[32 + d0 + j]      * scale);
            aQ1[8 + j] = (_Float16)(qrow[32 + d0 + 8 + j] * scale);
        }
    }

    // ---- Pass 1: scores = (Q*f/T) @ K^T, masked, into LDS ----
    for (int t0 = 0; t0 < S_LEN; t0 += 16) {
        // B operand, contiguous k order used here (matches A): lane n = lane&15; elem j -> K = 16*hi + j
        const float* krow = kh + (size_t)(t0 + nIdx) * D_DIM;
        if (t0 + 16 < S_LEN)
            __builtin_prefetch(kh + (size_t)(t0 + 16 + nIdx) * D_DIM, 0, 1);
        v16h bK0, bK1;
        {
            const int d0 = 16 * hi;
            const v4f* p0 = (const v4f*)(krow + d0);
            const v4f* p1 = (const v4f*)(krow + 32 + d0);
            #pragma unroll
            for (int q4 = 0; q4 < 4; ++q4) {
                v4f x0 = p0[q4], x1 = p1[q4];
                #pragma unroll
                for (int e = 0; e < 4; ++e) {
                    bK0[4 * q4 + e] = (_Float16)x0[e];
                    bK1[4 * q4 + e] = (_Float16)x1[e];
                }
            }
        }
        v8f c8 = {};
        c8 = __builtin_amdgcn_wmma_f32_16x16x32_f16(false, aQ0, false, bK0, (short)0, c8, false, false);
        c8 = __builtin_amdgcn_wmma_f32_16x16x32_f16(false, aQ1, false, bK1, (short)0, c8, false, false);

        // C layout: VGPR r -> M = r + 8*hi ; N = lane&15
        const int t = t0 + nIdx;
        #pragma unroll
        for (int r = 0; r < 8; ++r) {
            const int m = r + 8 * hi;
            const int s = row0 + m;
            float sc = c8[r];
            sc = (mh[(size_t)s * S_LEN + t] == 0) ? NEG_BIG : sc;
            sP[m * ROW_PAD + t] = sc;
        }
    }
    __syncthreads();

    // ---- Softmax over LDS rows: lane = (row = lane&15, half = hi covers 512 cols) ----
    const int srow = nIdx;
    float* rowp = sP + srow * ROW_PAD + hi * 512;

    float mx = -3.0e38f;
    for (int i = 0; i < 512; i += 4) {
        v4f x = *(const v4f*)(rowp + i);
        mx = fmaxf(mx, fmaxf(fmaxf(x[0], x[1]), fmaxf(x[2], x[3])));
    }
    sRed[lane] = mx;
    __syncthreads();
    mx = fmaxf(sRed[srow], sRed[srow + 16]);

    float sum = 0.0f;
    for (int i = 0; i < 512; i += 4) {
        v4f x = *(const v4f*)(rowp + i);
        v4f e;
        e[0] = __expf(x[0] - mx);
        e[1] = __expf(x[1] - mx);
        e[2] = __expf(x[2] - mx);
        e[3] = __expf(x[3] - mx);
        sum += (e[0] + e[1]) + (e[2] + e[3]);
        *(v4f*)(rowp + i) = e;
    }
    sRed[lane] = sum;
    __syncthreads();
    const float inv = 1.0f / (sRed[srow] + sRed[srow + 16]);
    if (hi == 0) sInv[srow] = inv;
    __syncthreads();

    // ---- Stream normalized attn to global: cooperative, coalesced, NON-TEMPORAL ----
    // attn is written once and never re-read: NT stores keep the 256MB stream out of
    // L2 so K/V (heavily re-read across the 64 row-tile workgroups per head) stay hot.
    #pragma unroll 1
    for (int m = 0; m < 16; ++m) {
        const float invm = sInv[m];
        float* grow = attnh + (size_t)(row0 + m) * S_LEN;
        const float* lrow = sP + m * ROW_PAD;
        for (int i = lane * 4; i < S_LEN; i += 128) {
            v4f e = *(const v4f*)(lrow + i);
            v4f r = e * invm;
            __builtin_nontemporal_store(r, (v4f*)(grow + i));
        }
    }

    // ---- Pass 2: out = P @ V (stage V tiles in LDS, f16 WMMA, 4 N-tiles of 16) ----
    v8f acc[4] = {};
    for (int t0 = 0; t0 < S_LEN; t0 += 32) {
        // stage V[t0..t0+31][0..63] coalesced: each lane copies one row
        {
            const v4f* src = (const v4f*)(vh + (size_t)(t0 + lane) * D_DIM);
            v4f* dst = (v4f*)(sV + lane * D_DIM);
            #pragma unroll
            for (int j = 0; j < 16; ++j) dst[j] = src[j];
        }
        __syncthreads();

        // A fragment of P: m = lane&15; elem j -> t = t0 + 16*hi + j
        v16h aP;
        {
            const float invm = sInv[nIdx];
            const float* pr = sP + nIdx * ROW_PAD + t0 + 16 * hi;
            #pragma unroll
            for (int j = 0; j < 8; ++j) {
                aP[j]     = (_Float16)(pr[j] * invm);
                aP[8 + j] = (_Float16)(pr[8 + j] * invm);
            }
        }
        #pragma unroll
        for (int dt = 0; dt < 4; ++dt) {
            // B fragment of V: n = lane&15 -> d = dt*16 + n ; elem j -> t_local = 16*hi + j
            v16h bV;
            const float* col = sV + (16 * hi) * D_DIM + dt * 16 + nIdx;
            #pragma unroll
            for (int j = 0; j < 16; ++j) bV[j] = (_Float16)col[j * D_DIM];
            acc[dt] = __builtin_amdgcn_wmma_f32_16x16x32_f16(false, aP, false, bV, (short)0, acc[dt], false, false);
        }
        __syncthreads();
    }

    // ---- Write output (also write-once -> NT): element r -> s = row0 + r + 8*hi, d = dt*16 + (lane&15)
    #pragma unroll
    for (int dt = 0; dt < 4; ++dt) {
        #pragma unroll
        for (int r = 0; r < 8; ++r) {
            __builtin_nontemporal_store(
                acc[dt][r],
                outh + (size_t)(row0 + r + 8 * hi) * D_DIM + dt * 16 + nIdx);
        }
    }
}

extern "C" void kernel_launch(void* const* d_in, const int* in_sizes, int n_in,
                              void* d_out, int out_size, void* d_ws, size_t ws_size,
                              hipStream_t stream) {
    const float* q      = (const float*)d_in[0];
    const float* k      = (const float*)d_in[1];
    const float* v      = (const float*)d_in[2];
    const float* factor = (const float*)d_in[3];
    const int*   mask   = (const int*)d_in[4];

    float* out  = (float*)d_out;
    float* attn = out + (size_t)4 * 16 * 1024 * 64;   // output first, then attn

    const size_t smem = (size_t)(16 * ROW_PAD + 32 * 64 + 32 + 16) * sizeof(float);
    dim3 grid(S_LEN / 16, 4 * 16);   // 64 row-tiles x 64 (B*H) heads
    dim3 block(32);                  // one wave32 per workgroup (EXEC all-ones for WMMA)
    hipLaunchKernelGGL(sdpa_wmma_kernel, grid, block, smem, stream,
                       q, k, v, factor, mask, out, attn);
}
